// SelfAttention_52209622450749
// MI455X (gfx1250) — matrix operand order repaired and checked by reading, not compiled, not run on hardware
//
#include <hip/hip_runtime.h>
#include <math.h>

// Problem constants (match reference)
#define SEQ      2048
#define HID      2048
#define NHEADS   16
#define HEAD_DIM 128
#define BATCH    2
#define ROWS     (BATCH * SEQ)   // 4096
#define NQKV     (3 * HID)       // 6144

typedef __attribute__((ext_vector_type(16))) __bf16 bf16x16;
typedef __attribute__((ext_vector_type(8)))  __bf16 bf16x8;
typedef __attribute__((ext_vector_type(8)))  float  f32x8;

__device__ __forceinline__ bf16x16 cat16(bf16x8 lo, bf16x8 hi) {
  bf16x16 r;
#pragma unroll
  for (int i = 0; i < 8; ++i) { r[i] = lo[i]; r[i + 8] = hi[i]; }
  return r;
}

// ---------------------------------------------------------------------------
// fp32 -> bf16 conversion
// ---------------------------------------------------------------------------
__global__ void cvt_f32_to_bf16(const float* __restrict__ src,
                                __bf16* __restrict__ dst, int n) {
  int i = blockIdx.x * blockDim.x + threadIdx.x;
  if (i < n) dst[i] = (__bf16)src[i];
}

// ---------------------------------------------------------------------------
// Tiled bf16 WMMA GEMM:  C[ROWS x N] = A[ROWS x 2048] * W[2048 x N] + bias
// Block: 256 threads (8 waves), tile 128x128, K-step 32.
// A tile is staged with GLOBAL_LOAD_ASYNC_TO_LDS_B128 (ASYNCcnt-tracked,
// bypasses VGPRs); W tile is staged transposed element-wise (2B scatter).
// MODE 0: scatter epilogue -> Q[bh][s][d], K[bh][s][d], Vt[bh][d][s] (bf16)
// MODE 1: plain fp32 store -> outF[row*N + col]
// ---------------------------------------------------------------------------
template<int MODE>
__global__ __launch_bounds__(256) void gemm_bf16(
    const __bf16* __restrict__ A,    // [ROWS][2048] row-major bf16
    const __bf16* __restrict__ W,    // [2048][N]    row-major bf16
    const float*  __restrict__ bias, // [N]
    int N,
    float*  __restrict__ outF,       // MODE 1
    __bf16* __restrict__ Qb,         // MODE 0
    __bf16* __restrict__ Kb,
    __bf16* __restrict__ Vt)
{
  __shared__ __bf16 As[128][48];   // [m][k], padded -> 96B row stride (16B aligned chunks)
  __shared__ __bf16 Bs[128][48];   // W transposed: [n][k]

  const int tid  = threadIdx.x;
  const int lane = tid & 31;
  const int wid  = tid >> 5;
  const int l15  = lane & 15;
  const int hh   = lane >> 4;              // which 16-lane half
  const int waveM = wid >> 1;              // 0..3 -> 32-row strip
  const int waveN = wid & 1;               // 0..1 -> 64-col strip
  const int tileM = blockIdx.y * 128;
  const int tileN = blockIdx.x * 128;

  f32x8 acc[2][4] = {};

  // cooperative staging indices
  const int aRow = tid >> 1;               // 0..127
  const int aK   = (tid & 1) * 16;         // 0,16
  const int bK   = tid >> 3;               // 0..31
  const int bN   = (tid & 7) * 16;         // 0..112

  // per-thread LDS destination for the async A-tile copy (32B contiguous)
  const unsigned ldsA = (unsigned)(size_t)&As[aRow][aK];

  for (int kb = 0; kb < 2048 / 32; ++kb) {
    // ---- stage A tile (128 x 32 bf16) via async global->LDS DMA ----
    const __bf16* ag = A + (size_t)(tileM + aRow) * 2048 + kb * 32 + aK;
    {
      const unsigned long long ga = (unsigned long long)(size_t)ag;
      // INST_OFFSET is added to BOTH the global and the LDS address, so one
      // base pair covers both 16B halves of this thread's 32B slice.
      asm volatile(
          "global_load_async_to_lds_b128 %0, %1, off\n\t"
          "global_load_async_to_lds_b128 %0, %1, off offset:16"
          :: "v"(ldsA), "v"(ga) : "memory");
    }
    __builtin_prefetch(ag + 32, 0, 1);     // next K tile -> global_prefetch_b8

    // ---- stage W tile transposed (32 x 128 -> Bs[n][k]) ----
    const __bf16* wg = W + (size_t)(kb * 32 + bK) * N + tileN + bN;
    bf16x8 w0 = *(const bf16x8*)(wg);
    bf16x8 w1 = *(const bf16x8*)(wg + 8);
    __builtin_prefetch(wg + (size_t)32 * N, 0, 1);
#pragma unroll
    for (int i = 0; i < 8; ++i) Bs[bN + i][bK]     = w0[i];
#pragma unroll
    for (int i = 0; i < 8; ++i) Bs[bN + 8 + i][bK] = w1[i];

    // drain this wave's async LDS writes, then block-wide barrier
    asm volatile("s_wait_asynccnt 0x0" ::: "memory");
    __syncthreads();

    // ---- WMMA: each wave 32x64 ----
    bf16x16 bfrag[4];
#pragma unroll
    for (int nt = 0; nt < 4; ++nt) {
      const int n = waveN * 64 + nt * 16 + l15;
      bfrag[nt] = cat16(*(const bf16x8*)&Bs[n][hh * 8],
                        *(const bf16x8*)&Bs[n][16 + hh * 8]);
    }
#pragma unroll
    for (int mt = 0; mt < 2; ++mt) {
      const int m = waveM * 32 + mt * 16 + l15;
      bf16x16 afrag = cat16(*(const bf16x8*)&As[m][hh * 8],
                            *(const bf16x8*)&As[m][16 + hh * 8]);
#pragma unroll
      for (int nt = 0; nt < 4; ++nt) {
        acc[mt][nt] = __builtin_amdgcn_wmma_f32_16x16x32_bf16(
            false, afrag, false, bfrag[nt], (short)0, acc[mt][nt],
            false, false);
      }
    }
    __syncthreads();
  }

  // ---- epilogue ----
#pragma unroll
  for (int mt = 0; mt < 2; ++mt) {
#pragma unroll
    for (int nt = 0; nt < 4; ++nt) {
      const int col = tileN + waveN * 64 + nt * 16 + l15;
      const float bv = bias[col];
#pragma unroll
      for (int r = 0; r < 8; ++r) {
        const int row = tileM + waveM * 32 + mt * 16 + r + 8 * hh;
        const float v = acc[mt][nt][r] + bv;
        if (MODE == 1) {
          outF[(size_t)row * N + col] = v;
        } else {
          const int sect  = col >> 11;          // 0=Q 1=K 2=V (uniform per nt)
          const int inner = col & 2047;
          const int head  = inner >> 7;
          const int d     = inner & 127;
          const int b     = row >> 11;
          const int s     = row & 2047;
          const int bh    = b * NHEADS + head;
          const __bf16 bw = (__bf16)v;
          if (sect == 0)
            Qb[((size_t)bh * SEQ + s) * HEAD_DIM + d] = bw;
          else if (sect == 1)
            Kb[((size_t)bh * SEQ + s) * HEAD_DIM + d] = bw;
          else
            Vt[((size_t)bh * HEAD_DIM + d) * SEQ + s] = bw;
        }
      }
    }
  }
}

// ---------------------------------------------------------------------------
// Flash attention, one wave (32 threads) per (16-query tile, batch*head).
// Logits = (q.k)/sqrt(128) with causal mask (equivalent to reference path).
// Q feeds the A operand directly; K (row-major) and Vt ([d][s]) feed the
// B operand with contiguous 16B loads. P is bounced C-layout->A-layout via LDS.
// ---------------------------------------------------------------------------
__global__ __launch_bounds__(32) void attn_kernel(
    const __bf16* __restrict__ Qb, const __bf16* __restrict__ Kb,
    const __bf16* __restrict__ Vt, __bf16* __restrict__ ctx)
{
  __shared__ __bf16 Plds[16][32];

  const int lane = threadIdx.x & 31;
  const int l15  = lane & 15;
  const int hh   = lane >> 4;
  const int qt   = blockIdx.x;     // 0..127
  const int bh   = blockIdx.y;     // 0..31
  const int q0   = qt * 16;

  const __bf16* Qbase = Qb + (size_t)bh * SEQ * HEAD_DIM;
  const __bf16* Kbase = Kb + (size_t)bh * SEQ * HEAD_DIM;
  const __bf16* Vbase = Vt + (size_t)bh * HEAD_DIM * SEQ;

  // Q tile as four 16x32 A-operand fragments
  bf16x16 qf[4];
  {
    const __bf16* qrow = Qbase + (size_t)(q0 + l15) * HEAD_DIM;
#pragma unroll
    for (int c = 0; c < 4; ++c)
      qf[c] = cat16(*(const bf16x8*)(qrow + c * 32 + hh * 8),
                    *(const bf16x8*)(qrow + c * 32 + 16 + hh * 8));
  }

  f32x8 accO[8] = {};
  float mrun[8], lrun[8];
#pragma unroll
  for (int r = 0; r < 8; ++r) { mrun[r] = -3.0e38f; lrun[r] = 0.0f; }

  const float LSCALE = 0.08838834764831845f;   // 8 * 1/(8*sqrt(128)) = 1/sqrt(128)
  const int nb = ((q0 + 15) >> 5) + 1;         // 32-key blocks touching causal window

  for (int kb = 0; kb < nb; ++kb) {
    // ---- S = Q * K^T for 32 keys (two 16-col WMMA tiles) ----
    f32x8 sacc[2] = {};
#pragma unroll
    for (int nt = 0; nt < 2; ++nt) {
      const int key = kb * 32 + nt * 16 + l15;
      const __bf16* krow = Kbase + (size_t)key * HEAD_DIM;
#pragma unroll
      for (int c = 0; c < 4; ++c) {
        bf16x16 kf = cat16(*(const bf16x8*)(krow + c * 32 + hh * 8),
                           *(const bf16x8*)(krow + c * 32 + 16 + hh * 8));
        sacc[nt] = __builtin_amdgcn_wmma_f32_16x16x32_bf16(
            false, qf[c], false, kf, (short)0, sacc[nt], false, false);
      }
    }

    // ---- online softmax over the 32 columns ----
    float pr[2][8];
    float alpha[8];
#pragma unroll
    for (int r = 0; r < 8; ++r) {
      const int rowg = q0 + r + 8 * hh;
      const int key0 = kb * 32 + l15;
      const int key1 = kb * 32 + 16 + l15;
      float L0 = (key0 <= rowg) ? sacc[0][r] * LSCALE : -3.0e38f;
      float L1 = (key1 <= rowg) ? sacc[1][r] * LSCALE : -3.0e38f;
      float mx = fmaxf(L0, L1);
#pragma unroll
      for (int off = 1; off < 16; off <<= 1)
        mx = fmaxf(mx, __shfl_xor(mx, off, 16));
      const float mnew = fmaxf(mrun[r], mx);
      const float p0 = __expf(L0 - mnew);
      const float p1 = __expf(L1 - mnew);
      float s = p0 + p1;
#pragma unroll
      for (int off = 1; off < 16; off <<= 1)
        s += __shfl_xor(s, off, 16);
      alpha[r] = __expf(mrun[r] - mnew);
      lrun[r]  = lrun[r] * alpha[r] + s;
      mrun[r]  = mnew;
      pr[0][r] = p0;
      pr[1][r] = p1;
    }
#pragma unroll
    for (int nf = 0; nf < 8; ++nf)
#pragma unroll
      for (int r = 0; r < 8; ++r) accO[nf][r] *= alpha[r];

    // ---- P: C layout -> A-operand layout through LDS ----
    __syncthreads();   // single wave: orders vs. previous iter's reads
#pragma unroll
    for (int nt = 0; nt < 2; ++nt)
#pragma unroll
      for (int r = 0; r < 8; ++r)
        Plds[r + 8 * hh][nt * 16 + l15] = (__bf16)pr[nt][r];
    __syncthreads();

    bf16x16 pa = cat16(*(const bf16x8*)&Plds[l15][hh * 8],
                       *(const bf16x8*)&Plds[l15][16 + hh * 8]);

    // ---- O += P * V (Vt gives contiguous B-operand fragments) ----
#pragma unroll
    for (int nf = 0; nf < 8; ++nf) {
      const int d = nf * 16 + l15;
      const __bf16* vrow = Vbase + (size_t)d * SEQ + kb * 32;
      bf16x16 vf = cat16(*(const bf16x8*)(vrow + hh * 8),
                         *(const bf16x8*)(vrow + 16 + hh * 8));
      accO[nf] = __builtin_amdgcn_wmma_f32_16x16x32_bf16(
          false, pa, false, vf, (short)0, accO[nf], false, false);
    }
  }

  // ---- write ctx[b][s][head*128 + d] as bf16, ready for out-proj GEMM ----
  const int b = bh >> 4, head = bh & 15;
#pragma unroll
  for (int nf = 0; nf < 8; ++nf) {
#pragma unroll
    for (int r = 0; r < 8; ++r) {
      const int s = q0 + r + 8 * hh;
      const float o = accO[nf][r] / lrun[r];
      ctx[((size_t)(b * SEQ + s)) * HID + head * HEAD_DIM + nf * 16 + l15] =
          (__bf16)o;
    }
  }
}

// ---------------------------------------------------------------------------
// Launch
// ---------------------------------------------------------------------------
extern "C" void kernel_launch(void* const* d_in, const int* in_sizes, int n_in,
                              void* d_out, int out_size, void* d_ws, size_t ws_size,
                              hipStream_t stream) {
  const float* hidden = (const float*)d_in[0];
  // d_in[1] = ltor_mask: unused (causal mask is computed analytically)
  const float* Wqkv = (const float*)d_in[2];
  const float* bqkv = (const float*)d_in[3];
  const float* Wout = (const float*)d_in[4];
  const float* bout = (const float*)d_in[5];
  float* out = (float*)d_out;

  char* ws = (char*)d_ws;
  size_t off = 0;
  auto carve = [&](size_t bytes) -> void* {
    void* p = ws + off;
    off += (bytes + 255) & ~(size_t)255;
    return p;
  };
  __bf16* hbf = (__bf16*)carve((size_t)ROWS * HID * 2);       // 16 MB
  __bf16* wqb = (__bf16*)carve((size_t)HID * NQKV * 2);       // 24 MB
  __bf16* wob = (__bf16*)carve((size_t)HID * HID * 2);        //  8 MB
  __bf16* Qb  = (__bf16*)carve((size_t)ROWS * HID * 2);       // 16 MB
  __bf16* Kb  = (__bf16*)carve((size_t)ROWS * HID * 2);       // 16 MB
  __bf16* Vt  = (__bf16*)carve((size_t)ROWS * HID * 2);       // 16 MB
  __bf16* ctx = (__bf16*)carve((size_t)ROWS * HID * 2);       // 16 MB

  const int nh = ROWS * HID;        // 8.4M
  const int nq = HID * NQKV;        // 12.6M
  const int nw = HID * HID;         // 4.2M
  cvt_f32_to_bf16<<<(nh + 255) / 256, 256, 0, stream>>>(hidden, hbf, nh);
  cvt_f32_to_bf16<<<(nq + 255) / 256, 256, 0, stream>>>(Wqkv, wqb, nq);
  cvt_f32_to_bf16<<<(nw + 255) / 256, 256, 0, stream>>>(Wout, wob, nw);

  // QKV projection + head-split scatter (V stored transposed)
  gemm_bf16<0><<<dim3(NQKV / 128, ROWS / 128), 256, 0, stream>>>(
      hbf, wqb, bqkv, NQKV, nullptr, Qb, Kb, Vt);

  // Flash attention: 128 query tiles x 32 (batch*head) waves
  attn_kernel<<<dim3(SEQ / 16, BATCH * NHEADS), 32, 0, stream>>>(Qb, Kb, Vt, ctx);

  // Output projection -> fp32 result
  gemm_bf16<1><<<dim3(HID / 128, ROWS / 128), 256, 0, stream>>>(
      ctx, wob, bout, HID, out, nullptr, nullptr, nullptr);
}
